// NucleusSampler_42039139893626
// MI455X (gfx1250) — matrix operand order repaired and checked by reading, not compiled, not run on hardware
//
#include <hip/hip_runtime.h>
#include <stdint.h>

// Nucleus (top-p) sampler for MI455X / gfx1250.
// One workgroup (1024 threads = 32 wave32s, one full WGP) per row.
// Row data is register-resident (13 x float4 per thread, b128 loads);
// nucleus cutoff found by binary search on probability value (no sort);
// final categorical pick via block prefix scan whose inter-wave stage is
// two v_wmma_f32_16x16x32_f16 triangular-mask matmuls.

#define VOCAB    50257
#define NT       1024
#define NPER4    13          // ceil(VOCAB / (NT*4)) float4 chunks per thread
#define NWAVES   32
#define THRESH   0.9f
#define BS_ITERS 24

typedef __attribute__((ext_vector_type(4)))  float    v4f;
typedef __attribute__((ext_vector_type(16))) _Float16 v16h;
typedef __attribute__((ext_vector_type(8)))  float    v8f;

__device__ __forceinline__ float wave_sum(float v) {
#pragma unroll
    for (int off = 16; off > 0; off >>= 1) v += __shfl_xor(v, off, 32);
    return v;
}

__device__ __forceinline__ float wave_max(float v) {
#pragma unroll
    for (int off = 16; off > 0; off >>= 1) v = fmaxf(v, __shfl_xor(v, off, 32));
    return v;
}

// deterministic per-row uniform in [0,1) (splitmix64 -> 24-bit mantissa)
__device__ __forceinline__ float row_uniform(int row) {
    uint64_t z = (uint64_t)(row + 1) * 0x9E3779B97F4A7C15ull + 0xD1B54A32D192ED03ull;
    z ^= z >> 30; z *= 0xBF58476D1CE4E5B9ull;
    z ^= z >> 27; z *= 0x94D049BB133111EBull;
    z ^= z >> 31;
    uint32_t m24 = (uint32_t)(z >> 40);
    return (float)m24 * (1.0f / 16777216.0f);
}

__global__ __launch_bounds__(NT) void nucleus_sample_kernel(
    const float* __restrict__ logits, int* __restrict__ out, int rows)
{
    __shared__ float s_redf[2][NWAVES]; // parity-double-buffered reduction scratch
    __shared__ float s_bc[2];           // parity-double-buffered broadcast
    __shared__ float s_wt[NWAVES];      // wave totals (WMMA scan input)
    __shared__ float s_y[NWAVES];       // inclusive scan of wave totals (WMMA out)
    __shared__ int   s_sel;             // winning thread id

    const int row  = blockIdx.x;
    if (row >= rows) return;            // block-uniform
    const int t    = threadIdx.x;
    const int lane = t & 31;
    const int wid  = t >> 5;
    const float* __restrict__ x = logits + (size_t)row * VOCAB;

    // ------------- Phase 0: streaming b128 load into registers -----------
    // element index of p[k][j] is (k*NT + t)*4 + j  (coalesced b128 per wave)
    v4f p[NPER4];
#pragma unroll
    for (int k = 0; k < NPER4 - 1; ++k) {
        const int base = (k * NT + t) * 4;          // base+3 <= 49151 < VOCAB
        p[k] = __builtin_nontemporal_load((const v4f*)(x + base));
    }
    {
        const int base = ((NPER4 - 1) * NT + t) * 4;
        v4f v;
        if (base + 3 < VOCAB) {
            v = __builtin_nontemporal_load((const v4f*)(x + base));
        } else {
#pragma unroll
            for (int j = 0; j < 4; ++j)
                v[j] = (base + j < VOCAB) ? x[base + j] : -__builtin_inff();
        }
        p[NPER4 - 1] = v;
    }

    // ------------- Phase 1: row max (block reduce, buffer 0) --------------
    float m = -__builtin_inff();
#pragma unroll
    for (int k = 0; k < NPER4; ++k)
#pragma unroll
        for (int j = 0; j < 4; ++j) m = fmaxf(m, p[k][j]);
    m = wave_max(m);
    if (lane == 0) s_redf[0][wid] = m;
    __syncthreads();
    if (wid == 0) {
        float v = wave_max(s_redf[0][lane]);
        if (lane == 0) s_bc[0] = v;
    }
    __syncthreads();
    m = s_bc[0];

    // ------------- Phase 2: exp + Z (block reduce, buffer 1) --------------
    float z = 0.0f;
#pragma unroll
    for (int k = 0; k < NPER4; ++k)
#pragma unroll
        for (int j = 0; j < 4; ++j) { p[k][j] = __expf(p[k][j] - m); z += p[k][j]; }
    z = wave_sum(z);
    if (lane == 0) s_redf[1][wid] = z;
    __syncthreads();
    if (wid == 0) {
        float v = wave_sum(s_redf[1][lane]);
        if (lane == 0) s_bc[1] = v;
    }
    __syncthreads();
    z = s_bc[1];

    const float inv = 1.0f / z;          // == max prob (max element exp == 1)
#pragma unroll
    for (int k = 0; k < NPER4; ++k)
#pragma unroll
        for (int j = 0; j < 4; ++j) p[k][j] *= inv;

    // ------------- Phase 3: binary search for nucleus cutoff --------------
    // F(t) = sum of probs >= t is decreasing; nucleus = {p >= tau},
    // tau = sup{ t : F(t) >= THRESH }. Invariant: F(lo) >= THRESH > F(hi).
    // Two barriers per iteration: scratch is parity-double-buffered.
    float lo = 0.0f;
    float hi = inv * 1.0000002f + 1e-35f;
    for (int it = 0; it < BS_ITERS; ++it) {
        const int   par = it & 1;
        const float tm  = 0.5f * (lo + hi);
        float s = 0.0f;
#pragma unroll
        for (int k = 0; k < NPER4; ++k)
#pragma unroll
            for (int j = 0; j < 4; ++j) s += (p[k][j] >= tm) ? p[k][j] : 0.0f;
        s = wave_sum(s);
        if (lane == 0) s_redf[par][wid] = s;
        __syncthreads();
        if (wid == 0) {
            float v = wave_sum(s_redf[par][lane]);
            if (lane == 0) s_bc[par] = v;
        }
        __syncthreads();
        const float F = s_bc[par];
        if (F >= THRESH) lo = tm; else hi = tm;
    }
    const float tau = lo;

    // ------------- Phase 4: masked local sum + in-wave scan ---------------
    float sloc = 0.0f;
#pragma unroll
    for (int k = 0; k < NPER4; ++k)
#pragma unroll
        for (int j = 0; j < 4; ++j) sloc += (p[k][j] >= tau) ? p[k][j] : 0.0f;

    float incl = sloc;                   // inclusive scan within wave (Kogge-Stone)
#pragma unroll
    for (int off = 1; off < 32; off <<= 1) {
        float v = __shfl_up(incl, off, 32);
        if (lane >= off) incl += v;
    }
    if (lane == 31) s_wt[wid] = incl;    // wave total
    __syncthreads();

    // ------------- Phase 5: inter-wave scan via WMMA (wave 0) -------------
    // y_m = sum_{k<=m} w_k and y_{m+16} = sum_{k<=m+16} w_k using two
    // v_wmma_f32_16x16x32_f16 with lower-triangular f16 A masks and
    // B[k][n] = w_k replicated over n.
    if (wid == 0) {
        const int base = (lane < 16) ? 0 : 16;  // B: lanes0-15 K=0..15, lanes16-31 K=16..31
        v16h bmat;
#pragma unroll
        for (int i = 0; i < 16; ++i) bmat[i] = (_Float16)s_wt[base + i];

        const int M    = lane & 15;             // A: lane%16 = row M
        const int koff = (lane < 16) ? 0 : 16;  // lanes0-15: K 0..15; lanes16-31: K 16..31
        v16h a0, a1;
#pragma unroll
        for (int i = 0; i < 16; ++i) {
            const int K = i + koff;
            a0[i] = (_Float16)((K <= M)      ? 1.0f : 0.0f);
            a1[i] = (_Float16)((K <= M + 16) ? 1.0f : 0.0f);
        }
        v8f c0 = {};
        v8f d0 = __builtin_amdgcn_wmma_f32_16x16x32_f16(false, a0, false, bmat,
                                                        (short)0, c0, false, false);
        v8f d1 = __builtin_amdgcn_wmma_f32_16x16x32_f16(false, a1, false, bmat,
                                                        (short)0, c0, false, false);
        // C/D layout: VGPR r -> M=r (lanes 0-15, N=lane) / M=r+8 (lanes 16-31)
        if (lane == 0) {
#pragma unroll
            for (int r = 0; r < 8; ++r) { s_y[r]     = d0[r]; s_y[16 + r] = d1[r]; }
        }
        if (lane == 16) {
#pragma unroll
            for (int r = 0; r < 8; ++r) { s_y[8 + r] = d0[r]; s_y[24 + r] = d1[r]; }
        }
    }
    __syncthreads();

    // ------------- Phase 6: categorical pick ------------------------------
    const float total = s_y[NWAVES - 1];                 // masked mass
    float u = row_uniform(row) * total;
    u = fminf(u, total * 0.99999994f);

    const float wbase = (wid == 0) ? 0.0f : s_y[wid - 1];
    const float tincl = wbase + incl;                    // inclusive bound
    const bool  fired = (sloc > 0.0f) && (u < tincl);    // min-t winner = first crossing

    int cand = -1;
    if (fired) {
        float c = tincl - sloc;
        int last = -1;
#pragma unroll
        for (int k = 0; k < NPER4; ++k) {
#pragma unroll
            for (int j = 0; j < 4; ++j) {
                const float pv = p[k][j];
                if (pv >= tau) {
                    c += pv;
                    const int idx = (k * NT + t) * 4 + j;
                    if (cand < 0 && u < c) cand = idx;
                    last = idx;
                }
            }
        }
        if (cand < 0) cand = last;                       // fp-edge fallback
    }

    if (t == 0) s_sel = 0x7FFFFFFF;
    __syncthreads();
    if (fired) atomicMin(&s_sel, t);
    __syncthreads();
    if (t == s_sel) out[row] = cand;
    else if (s_sel == 0x7FFFFFFF && t == 0) out[row] = 0; // degenerate fallback
}

extern "C" void kernel_launch(void* const* d_in, const int* in_sizes, int n_in,
                              void* d_out, int out_size, void* d_ws, size_t ws_size,
                              hipStream_t stream)
{
    const float* logits = (const float*)d_in[0];
    const int rows = in_sizes[0] / VOCAB;                // 1024
    int* out = (int*)d_out;
    nucleus_sample_kernel<<<rows, NT, 0, stream>>>(logits, out, rows);
    (void)n_in; (void)out_size; (void)d_ws; (void)ws_size;
}
